// Block_44959717655131
// MI455X (gfx1250) — matrix operand order repaired and checked by reading, not compiled, not run on hardware
//
#include <hip/hip_runtime.h>
#include <hip/hip_bf16.h>

typedef _Float16 half_t;
typedef __attribute__((ext_vector_type(16))) _Float16 v16h;
typedef __attribute__((ext_vector_type(8)))  _Float16 v8h;
typedef __attribute__((ext_vector_type(2)))  _Float16 v2h;
typedef __attribute__((ext_vector_type(8)))  float    v8f;
typedef __attribute__((ext_vector_type(4)))  unsigned int u32x4;
typedef __attribute__((ext_vector_type(8)))  unsigned int u32x8;

static constexpr int kB  = 4;
static constexpr int kT  = 2048;
static constexpr int kD  = 1024;
static constexpr int kH  = 16;
static constexpr int kHS = 64;
static constexpr int kM  = kB * kT;    // 8192 rows
static constexpr int kDF = 4 * kD;     // 4096 ffn dim

__device__ __forceinline__ v16h mk16(v8h a, v8h b) {
  return __builtin_shufflevector(a, b, 0,1,2,3,4,5,6,7,8,9,10,11,12,13,14,15);
}

__device__ __forceinline__ v8f wmma16(v16h a, v16h b, v8f c) {
  // D(16x16,f32) = A(16x32,f16) * B(32x16,f16) + C
  return __builtin_amdgcn_wmma_f32_16x16x32_f16(false, a, false, b, (short)0, c,
                                                false, false);
}

// async global->LDS copy, 16B per lane; the offset field adds to both LDS and global address
__device__ __forceinline__ void async_copy_b128(unsigned lds_byte_off, const void* g) {
  asm volatile("global_load_async_to_lds_b128 %0, %1, off"
               :: "v"(lds_byte_off), "v"(g) : "memory");
}
__device__ __forceinline__ void async_copy_b128_o16(unsigned lds_byte_off, const void* g) {
  asm volatile("global_load_async_to_lds_b128 %0, %1, off offset:16"
               :: "v"(lds_byte_off), "v"(g) : "memory");
}
__device__ __forceinline__ void wait_asynccnt0() {
  asm volatile("s_wait_asynccnt 0x0" ::: "memory");
}
__device__ __forceinline__ void wait_tensorcnt0() {
  __builtin_amdgcn_s_wait_tensorcnt(0);
  asm volatile("" ::: "memory");
}

// TDM: 2-D tile load global->LDS. D# per CDNA5 ISA ch.8:
//  group0: [1:0]count=1 | [63:32]lds_addr | [120:64]global_addr | [127:126]type=2
//  group1: data_size=2B, pad 16DW + 4DW (row 32 halfs -> stride 40 halfs),
//          tensor_dim0/1 = tile dims (all in-bounds), tensor_dim0_stride = row stride.
__device__ __forceinline__ void tdm_load_2d_a(unsigned lds_byte_off, const void* gaddr,
                                              unsigned stride_units) {
  unsigned long long ga = (unsigned long long)gaddr;
  u32x4 g0;
  g0[0] = 1u;                                            // count=1, user mode
  g0[1] = lds_byte_off;                                  // lds_addr
  g0[2] = (unsigned)ga;                                  // global_addr[31:0]
  g0[3] = (unsigned)((ga >> 32) & 0x1FFFFFFu) | (2u << 30);  // addr[56:32] | type=2
  u32x8 g1;
  g1[0] = (1u << 16) | (1u << 20) | (3u << 22) | (3u << 25); // 2B, pad_en, 16DW, +4DW
  g1[1] = (32u & 0xFFFFu) << 16;                         // tensor_dim0=32 (lo16 in hi half)
  g1[2] = (32u >> 16) | ((128u & 0xFFFFu) << 16);        // tensor_dim0 hi | tensor_dim1 lo
  g1[3] = (128u >> 16) | (32u << 16);                    // tensor_dim1 hi | tile_dim0=32
  g1[4] = 128u;                                          // tile_dim1=128, tile_dim2=0
  g1[5] = stride_units;                                  // tensor_dim0_stride[31:0]
  g1[6] = 0u;
  g1[7] = 0u;
  asm volatile("tensor_load_to_lds %0, %1" :: "s"(g0), "s"(g1) : "memory");
}

// ---------------------------------------------------------------- cast f32->f16
__global__ __launch_bounds__(256)
void cast_kernel(const float* __restrict__ src, half_t* __restrict__ dst, size_t n) {
  size_t i = (size_t)blockIdx.x * 256 + threadIdx.x;
  if (i < n) dst[i] = (half_t)src[i];
}

// ---------------------------------------------------------------- layernorm -> f16
__global__ __launch_bounds__(256)
void ln_kernel(const float* __restrict__ x, const float* __restrict__ g,
               const float* __restrict__ bt, half_t* __restrict__ outh) {
  __shared__ float red[256];
  const int row = blockIdx.x;
  const float* xr = x + (size_t)row * kD;
  float s = 0.f, s2 = 0.f;
  for (int i = threadIdx.x; i < kD; i += 256) { float v = xr[i]; s += v; s2 += v * v; }
  red[threadIdx.x] = s; __syncthreads();
  for (int st = 128; st > 0; st >>= 1) {
    if (threadIdx.x < st) red[threadIdx.x] += red[threadIdx.x + st];
    __syncthreads();
  }
  float S = red[0]; __syncthreads();
  red[threadIdx.x] = s2; __syncthreads();
  for (int st = 128; st > 0; st >>= 1) {
    if (threadIdx.x < st) red[threadIdx.x] += red[threadIdx.x + st];
    __syncthreads();
  }
  float S2 = red[0];
  float mu  = S * (1.f / kD);
  float var = S2 * (1.f / kD) - mu * mu;
  float rs  = rsqrtf(var + 1e-5f);
  for (int i = threadIdx.x; i < kD; i += 256)
    outh[(size_t)row * kD + i] = (half_t)((xr[i] - mu) * rs * g[i] + bt[i]);
}

// ------------------------------------------------ V transpose via LDS 64x64 tiles
__global__ __launch_bounds__(256)
void transpose_v_kernel(const half_t* __restrict__ vh, half_t* __restrict__ vt) {
  __shared__ __align__(16) half_t tile[64][72];
  const int tid = threadIdx.x;
  const int bh  = blockIdx.y;          // b*H + h
  const int t0  = blockIdx.x * 64;
  const int b   = bh >> 4, h = bh & 15;
  const int i   = tid >> 2;            // 0..63
  const int j0  = (tid & 3) * 16;      // 0,16,32,48
  const half_t* src = vh + ((size_t)(b * kT + t0 + i)) * kD + h * kHS + j0;
  *(v8h*)(&tile[i][j0])     = *(const v8h*)src;
  *(v8h*)(&tile[i][j0 + 8]) = *(const v8h*)(src + 8);
  __syncthreads();
  v8h r0, r1;
#pragma unroll
  for (int k = 0; k < 8; ++k) r0[k] = tile[j0 + k][i];
#pragma unroll
  for (int k = 0; k < 8; ++k) r1[k] = tile[j0 + 8 + k][i];
  half_t* dst = vt + ((size_t)bh * kHS + i) * kT + t0 + j0;
  *(v8h*)dst       = r0;
  *(v8h*)(dst + 8) = r1;
}

// ---------------------------------------------------------------- tiled WMMA GEMM
// C[M,N] = A[M,K]*B[K,N] (f16 in, f32 acc), 128x128 tile / 256-thread block.
// A tile staged by TDM (tensor_load_to_lds, padded rows), B tile transposed with
// paired ds_store_b32; double-buffered, TDM latency overlapped with WMMA.
template <bool OUT_F16, bool RELU, bool BIAS, bool RES>
__global__ __launch_bounds__(256)
void gemm_kernel(const half_t* __restrict__ A, const half_t* __restrict__ B,
                 half_t* __restrict__ outh, float* __restrict__ outf,
                 const float* __restrict__ bias, const float* __restrict__ res,
                 int M, int N, int K) {
  constexpr int LDSW = 40;                 // padded row stride in halfs (80B)
  constexpr int HALF_BUF = 2 * 128 * LDSW; // As + Bt per buffer (halfs)
  __shared__ __align__(16) half_t smem[2 * HALF_BUF];  // 40KB, LDS base 0

  const int tid  = threadIdx.x;
  const int lane = tid & 31;
  const int w    = tid >> 5;
  const int wm   = w >> 2;
  const int wn   = w & 3;
  const int hi   = lane >> 4;
  const int l15  = lane & 15;
  const int m0   = blockIdx.y * 128;
  const int n0   = blockIdx.x * 128;

  const int bpr  = tid >> 4;            // B: k-row pair 0..15
  const int bnp  = (tid & 15) * 8;      // B: n offset 0..120

  v8f zero = {0.f,0.f,0.f,0.f,0.f,0.f,0.f,0.f};
  v8f acc[4][2];
#pragma unroll
  for (int i = 0; i < 4; ++i)
#pragma unroll
    for (int j = 0; j < 2; ++j) acc[i][j] = zero;

  const int KT = K >> 5;

  // ---- prologue: stage tile 0 into buffer 0
  if (w == 0)
    tdm_load_2d_a(0u, A + (size_t)m0 * K, (unsigned)K);
  v8h bx0, bx1;
  {
    const half_t* g0 = B + (size_t)(2 * bpr) * N + n0 + bnp;
    bx0 = *(const v8h*)g0;
    bx1 = *(const v8h*)(g0 + N);
  }
  {
    half_t* Bt = smem + 128 * LDSW;
#pragma unroll
    for (int j = 0; j < 8; ++j) {
      v2h p = {bx0[j], bx1[j]};
      *(v2h*)(Bt + (bnp + j) * LDSW + 2 * bpr) = p;
    }
  }
  if (w == 0) wait_tensorcnt0();
  __syncthreads();

  for (int kt = 0; kt < KT; ++kt) {
    const int cur = kt & 1;
    const int nxt = cur ^ 1;
    const bool has_next = (kt + 1 < KT);
    const half_t* As = smem + cur * HALF_BUF;
    const half_t* Bt = smem + cur * HALF_BUF + 128 * LDSW;

    if (has_next) {
      const int k0n = (kt + 1) * 32;
      if (w == 0)   // TDM copy of next A tile overlaps the WMMAs below
        tdm_load_2d_a((unsigned)(nxt * HALF_BUF * 2),
                      A + (size_t)m0 * K + k0n, (unsigned)K);
      const half_t* g0 = B + (size_t)(k0n + 2 * bpr) * N + n0 + bnp;
      bx0 = *(const v8h*)g0;
      bx1 = *(const v8h*)(g0 + N);
    }

    // ---- compute current tile
    v16h af[4];
#pragma unroll
    for (int i = 0; i < 4; ++i) {
      const half_t* p = As + (wm * 64 + i * 16 + l15) * LDSW;
      af[i] = mk16(*(const v8h*)(p + 8 * hi), *(const v8h*)(p + 16 + 8 * hi));
    }
    v16h bf[2];
#pragma unroll
    for (int j = 0; j < 2; ++j) {
      const half_t* p = Bt + (wn * 32 + j * 16 + l15) * LDSW;
      bf[j] = mk16(*(const v8h*)(p + 8 * hi), *(const v8h*)(p + 16 + 8 * hi));
    }
#pragma unroll
    for (int i = 0; i < 4; ++i)
#pragma unroll
      for (int j = 0; j < 2; ++j) acc[i][j] = wmma16(af[i], bf[j], acc[i][j]);

    if (has_next) {   // transpose-store next B tile into the other buffer
      half_t* Btn = smem + nxt * HALF_BUF + 128 * LDSW;
#pragma unroll
      for (int j = 0; j < 8; ++j) {
        v2h p = {bx0[j], bx1[j]};
        *(v2h*)(Btn + (bnp + j) * LDSW + 2 * bpr) = p;
      }
    }
    if (w == 0) wait_tensorcnt0();
    __syncthreads();
  }

  // epilogue
#pragma unroll
  for (int i = 0; i < 4; ++i)
#pragma unroll
    for (int j = 0; j < 2; ++j) {
      int rowb = m0 + wm * 64 + i * 16 + 8 * hi;
      int col  = n0 + wn * 32 + j * 16 + l15;
      float bv = BIAS ? bias[col] : 0.f;
#pragma unroll
      for (int r = 0; r < 8; ++r) {
        float v = acc[i][j][r] + bv;
        if (RES)  v += res[(size_t)(rowb + r) * N + col];
        if (RELU) v = fmaxf(v, 0.f);
        if (OUT_F16) outh[(size_t)(rowb + r) * N + col] = (half_t)v;
        else         outf[(size_t)(rowb + r) * N + col] = v;
      }
    }
}

// ---------------------------------------------------------------- causal attention
// grid (T/64, H, B), 128 threads = 4 waves; each wave owns 16 query rows.
// K chunks (32 kv rows x 64 halfs) staged via async-to-LDS, shared by all 4 waves,
// double-buffered; padded to 72 halfs/row for bank spread.
static constexpr int kKROW = 72;                 // padded K row (halfs)
static constexpr int kKBUF = 32 * kKROW;         // one K chunk (halfs) = 2304

__device__ __forceinline__ void stage_k_chunk(const half_t* Kb, int kv0,
                                              unsigned buf_byte_base, int tid) {
  const int row = tid >> 2;            // 0..31
  const int cph = (tid & 3) * 16;      // half offset in row: 0,16,32,48
  const half_t* g = Kb + (size_t)(kv0 + row) * kD + cph;
  unsigned loff = buf_byte_base + (unsigned)(row * (kKROW * 2) + cph * 2);
  async_copy_b128(loff, g);
  async_copy_b128_o16(loff, g);
}

__device__ __forceinline__ v8f qk_tile_lds(const half_t* Ks, int hv,
                                           int l15, int hi, v16h qa0, v16h qa1) {
  const half_t* krow = Ks + (hv * 16 + l15) * kKROW;
  v16h bk0 = mk16(*(const v8h*)(krow + 8 * hi),      *(const v8h*)(krow + 16 + 8 * hi));
  v16h bk1 = mk16(*(const v8h*)(krow + 32 + 8 * hi), *(const v8h*)(krow + 48 + 8 * hi));
  v8f s = {0.f,0.f,0.f,0.f,0.f,0.f,0.f,0.f};
  s = wmma16(qa0, bk0, s);
  s = wmma16(qa1, bk1, s);
  return s;
}

__global__ __launch_bounds__(128)
void attn_kernel(const half_t* __restrict__ qhp, const half_t* __restrict__ khp,
                 const half_t* __restrict__ vtp, half_t* __restrict__ op) {
  // [ Kbuf0 | Kbuf1 | Pt(4 waves x 16 x 40) ]  -- single object => LDS base 0
  __shared__ __align__(16) half_t att_smem[2 * kKBUF + 4 * 16 * 40];
  half_t* PtBase = att_smem + 2 * kKBUF;

  const int tid  = threadIdx.x;
  const int w    = tid >> 5;
  const int lane = tid & 31;
  const int hi   = lane >> 4;
  const int l15  = lane & 15;
  const int qblk = blockIdx.x;
  const int h    = blockIdx.y;
  const int b    = blockIdx.z;
  const int wq0  = qblk * 64 + w * 16;
  const float scale = 0.03125f;     // D^-0.5 = 1024^-0.5

  const half_t* Qb = qhp + ((size_t)(b * kT + wq0)) * kD + h * kHS;
  const half_t* Kb = khp + ((size_t)b * kT) * kD + h * kHS;
  const half_t* Vb = vtp + ((size_t)(b * kH + h)) * kHS * kT;   // [HS][T]

  const half_t* qrow = Qb + (size_t)l15 * kD;
  v16h qa0 = mk16(*(const v8h*)(qrow + 8 * hi),      *(const v8h*)(qrow + 16 + 8 * hi));
  v16h qa1 = mk16(*(const v8h*)(qrow + 32 + 8 * hi), *(const v8h*)(qrow + 48 + 8 * hi));

  const int nchunk = (qblk + 1) * 2;   // 32-col chunks, workgroup-uniform

  // ---- pass 1: per-row max
  float mrow[8];
#pragma unroll
  for (int r = 0; r < 8; ++r) mrow[r] = -3.0e38f;

  stage_k_chunk(Kb, 0, 0u, tid);
  for (int c = 0; c < nchunk; ++c) {
    const int cur = c & 1, nxt = cur ^ 1;
    wait_asynccnt0();
    __syncthreads();
    if (c + 1 < nchunk)
      stage_k_chunk(Kb, (c + 1) * 32, (unsigned)(nxt * kKBUF * 2), tid);
    const half_t* Ks = att_smem + cur * kKBUF;
#pragma unroll
    for (int hv = 0; hv < 2; ++hv) {
      v8f s = qk_tile_lds(Ks, hv, l15, hi, qa0, qa1);
      int col = c * 32 + hv * 16 + l15;
#pragma unroll
      for (int r = 0; r < 8; ++r) {
        int qg = wq0 + r + 8 * hi;
        float v = (col <= qg) ? s[r] * scale : -3.0e38f;
        mrow[r] = fmaxf(mrow[r], v);
      }
    }
  }
#pragma unroll
  for (int r = 0; r < 8; ++r) {
    float v = mrow[r];
    v = fmaxf(v, __shfl_xor(v, 1));
    v = fmaxf(v, __shfl_xor(v, 2));
    v = fmaxf(v, __shfl_xor(v, 4));
    v = fmaxf(v, __shfl_xor(v, 8));
    mrow[r] = v;
  }

  // ---- pass 2: P = exp(S-m); accumulate l and O = P*V
  float lrow[8];
#pragma unroll
  for (int r = 0; r < 8; ++r) lrow[r] = 0.f;
  v8f zero = {0.f,0.f,0.f,0.f,0.f,0.f,0.f,0.f};
  v8f o[4];
#pragma unroll
  for (int j = 0; j < 4; ++j) o[j] = zero;

  __syncthreads();                    // pass1 reads done before restaging buf0
  stage_k_chunk(Kb, 0, 0u, tid);
  for (int c = 0; c < nchunk; ++c) {
    const int cur = c & 1, nxt = cur ^ 1;
    wait_asynccnt0();
    __syncthreads();
    if (c + 1 < nchunk)
      stage_k_chunk(Kb, (c + 1) * 32, (unsigned)(nxt * kKBUF * 2), tid);
    const half_t* Ks = att_smem + cur * kKBUF;
#pragma unroll
    for (int hv = 0; hv < 2; ++hv) {
      v8f s = qk_tile_lds(Ks, hv, l15, hi, qa0, qa1);
      int col = c * 32 + hv * 16 + l15;
#pragma unroll
      for (int r = 0; r < 8; ++r) {
        int qg = wq0 + r + 8 * hi;
        float p = (col <= qg) ? __expf(s[r] * scale - mrow[r]) : 0.f;
        lrow[r] += p;
        PtBase[(w * 16 + r + 8 * hi) * 40 + hv * 16 + l15] = (half_t)p;
      }
    }
    asm volatile("s_wait_dscnt 0x0" ::: "memory");   // P tile visible to own wave
    const half_t* prow = PtBase + (w * 16 + l15) * 40;
    v16h pa = mk16(*(const v8h*)(prow + 8 * hi), *(const v8h*)(prow + 16 + 8 * hi));
#pragma unroll
    for (int j = 0; j < 4; ++j) {
      const half_t* vrow = Vb + (size_t)(j * 16 + l15) * kT + c * 32;
      v16h bv = mk16(*(const v8h*)(vrow + 8 * hi), *(const v8h*)(vrow + 16 + 8 * hi));
      o[j] = wmma16(pa, bv, o[j]);
    }
  }

#pragma unroll
  for (int r = 0; r < 8; ++r) {
    float v = lrow[r];
    v += __shfl_xor(v, 1);
    v += __shfl_xor(v, 2);
    v += __shfl_xor(v, 4);
    v += __shfl_xor(v, 8);
    lrow[r] = v;
  }
#pragma unroll
  for (int j = 0; j < 4; ++j)
#pragma unroll
    for (int r = 0; r < 8; ++r) {
      int t = wq0 + r + 8 * hi;
      op[((size_t)(b * kT + t)) * kD + h * kHS + j * 16 + l15] =
          (half_t)(o[j][r] / lrow[r]);
    }
}

// ---------------------------------------------------------------- host orchestration
extern "C" void kernel_launch(void* const* d_in, const int* in_sizes, int n_in,
                              void* d_out, int out_size, void* d_ws, size_t ws_size,
                              hipStream_t stream) {
  const float* x   = (const float*)d_in[0];
  const float* wq  = (const float*)d_in[1];
  const float* wk  = (const float*)d_in[2];
  const float* wv  = (const float*)d_in[3];
  const float* wo  = (const float*)d_in[4];
  const float* bo  = (const float*)d_in[5];
  const float* w1  = (const float*)d_in[6];
  const float* b1  = (const float*)d_in[7];
  const float* w2  = (const float*)d_in[8];
  const float* b2  = (const float*)d_in[9];
  const float* g1  = (const float*)d_in[10];
  const float* be1 = (const float*)d_in[11];
  const float* g2  = (const float*)d_in[12];
  const float* be2 = (const float*)d_in[13];
  float* out = (float*)d_out;

  char* ws = (char*)d_ws;
  const size_t MB = 1ull << 20;
  half_t* wqh   = (half_t*)(ws + 0 * MB);     // 2MB
  half_t* wkh   = (half_t*)(ws + 2 * MB);     // 2MB
  half_t* wvh   = (half_t*)(ws + 4 * MB);     // 2MB
  half_t* woh   = (half_t*)(ws + 6 * MB);     // 2MB
  half_t* w1h   = (half_t*)(ws + 8 * MB);     // 8MB
  half_t* w2h   = (half_t*)(ws + 16 * MB);    // 8MB
  half_t* xnh   = (half_t*)(ws + 24 * MB);    // 16MB (reused as xn2)
  half_t* qh    = (half_t*)(ws + 40 * MB);    // 16MB
  half_t* kh    = (half_t*)(ws + 56 * MB);    // 16MB
  half_t* vh    = (half_t*)(ws + 72 * MB);    // 16MB (reused as attn out)
  half_t* vth   = (half_t*)(ws + 88 * MB);    // 16MB
  float*  x2    = (float*)(ws + 104 * MB);    // 32MB
  half_t* h1    = (half_t*)(ws + 40 * MB);    // 64MB, reuses qh..vth (dead by then)
  half_t* attnh = vh;                         // total footprint: 136MB

  const size_t nDD = (size_t)kD * kD;
  const size_t nDF = (size_t)kD * kDF;

  // 1) cast weights to f16
  cast_kernel<<<(unsigned)((nDD + 255) / 256), 256, 0, stream>>>(wq, wqh, nDD);
  cast_kernel<<<(unsigned)((nDD + 255) / 256), 256, 0, stream>>>(wk, wkh, nDD);
  cast_kernel<<<(unsigned)((nDD + 255) / 256), 256, 0, stream>>>(wv, wvh, nDD);
  cast_kernel<<<(unsigned)((nDD + 255) / 256), 256, 0, stream>>>(wo, woh, nDD);
  cast_kernel<<<(unsigned)((nDF + 255) / 256), 256, 0, stream>>>(w1, w1h, nDF);
  cast_kernel<<<(unsigned)((nDF + 255) / 256), 256, 0, stream>>>(w2, w2h, nDF);

  // 2) LN1: x -> xn (f16)
  ln_kernel<<<kM, 256, 0, stream>>>(x, g1, be1, xnh);

  // 3) QKV projections (f16 out, no bias)
  dim3 gDD(kD / 128, kM / 128);
  gemm_kernel<true, false, false, false><<<gDD, 256, 0, stream>>>(
      xnh, wqh, qh, nullptr, nullptr, nullptr, kM, kD, kD);
  gemm_kernel<true, false, false, false><<<gDD, 256, 0, stream>>>(
      xnh, wkh, kh, nullptr, nullptr, nullptr, kM, kD, kD);
  gemm_kernel<true, false, false, false><<<gDD, 256, 0, stream>>>(
      xnh, wvh, vh, nullptr, nullptr, nullptr, kM, kD, kD);

  // 4) transpose V -> [B,H,HS,T] (LDS-tiled, coalesced)
  transpose_v_kernel<<<dim3(kT / 64, kB * kH), 256, 0, stream>>>(vh, vth);

  // 5) causal attention -> attnh (f16, [M,D] with heads interleaved)
  attn_kernel<<<dim3(kT / 64, kH, kB), 128, 0, stream>>>(qh, kh, vth, attnh);

  // 6) out-proj: x2 = x + attn @ wo + bo (f32)
  gemm_kernel<false, false, true, true><<<gDD, 256, 0, stream>>>(
      attnh, woh, nullptr, x2, bo, x, kM, kD, kD);

  // 7) LN2: x2 -> xn2 (f16, reuse xnh)
  ln_kernel<<<kM, 256, 0, stream>>>(x2, g2, be2, xnh);

  // 8) FFN1: h1 = relu(xn2 @ w1 + b1) (f16)
  dim3 gDF(kDF / 128, kM / 128);
  gemm_kernel<true, true, true, false><<<gDF, 256, 0, stream>>>(
      xnh, w1h, h1, nullptr, b1, nullptr, kM, kDF, kD);

  // 9) FFN2: out = x2 + h1 @ w2 + b2 (f32)
  gemm_kernel<false, false, true, true><<<gDD, 256, 0, stream>>>(
      h1, w2h, nullptr, out, b2, x2, kM, kD, kDF);
}
